// SingleMPNN_18124761989527
// MI455X (gfx1250) — compile-verified, matrix operand order checked by reading, not run on hardware
//
#include <hip/hip_runtime.h>
#include <hip/hip_bf16.h>

// ---------------------------------------------------------------------------
// Fused MPNN for MI455X (gfx1250), bf16 WMMA (v_wmma_f32_16x16x32_bf16).
//   edge kernel : gather -> 2x [32x256 @ 256x256] -> softplus -> atomic scatter
//                 (MT=2 edge tiles per wave: each B chunk feeds 2 WMMAs,
//                  halving weight re-fetch traffic vs 1 tile/wave)
//   node kernel : [x||agg] -> 384x256, 256x256, 256x128 -> out
// Messages never touch HBM; agg (51 MB) lives in the 192 MB L2.
// Softplus uses v_exp_f32 / v_log_f32.
// ---------------------------------------------------------------------------

#define DIN   128
#define DHID  256
#define DOUT  128
#define EDGES 800000
#define BNODES 50000
#define MT    2          // edge tiles (16 rows each) per wave

typedef __bf16 bf16_t;
typedef __attribute__((ext_vector_type(16))) __bf16 v16bf;
typedef __attribute__((ext_vector_type(8)))  __bf16 v8bf;
typedef __attribute__((ext_vector_type(8)))  float  v8f;

static __device__ inline float softplus_f(float v) {
  // max(x,0) + log1p(exp(-|x|)) with HW transcendentals (v_exp_f32/v_log_f32)
  return fmaxf(v, 0.0f) + __logf(1.0f + __expf(-fabsf(v)));
}

// CDNA5 16-bit operand layout (ISA 7.12.2): lane-group g (=lane/16), half h:
//   K(h,g) = 16*(h/8) + 8*g + (h%8)   -> two contiguous 8-half (16B) runs.
// Works for A (lane = M row) and B (lane = N col, reading W^T row-major).
template <typename P>
static __device__ inline v16bf load_bf_chunk(P base, int kbase, int g) {
  union { v16bf v; v8bf h[2]; } u;
  u.h[0] = *reinterpret_cast<const v8bf*>(&base[kbase + 8 * g]);
  u.h[1] = *reinterpret_cast<const v8bf*>(&base[kbase + 16 + 8 * g]);
  return u.v;
}

// Same layout but sourced from an f32 row (agg), converting to bf16.
static __device__ inline v16bf load_f32_chunk(const float* base, int kbase, int g) {
  union { v16bf v; __bf16 e[16]; } u;
  const float4* p0 = reinterpret_cast<const float4*>(base + kbase + 8 * g);
  const float4* p1 = reinterpret_cast<const float4*>(base + kbase + 16 + 8 * g);
  float4 f0 = p0[0], f1 = p0[1], f2 = p1[0], f3 = p1[1];
  u.e[0]  = (__bf16)f0.x; u.e[1]  = (__bf16)f0.y; u.e[2]  = (__bf16)f0.z; u.e[3]  = (__bf16)f0.w;
  u.e[4]  = (__bf16)f1.x; u.e[5]  = (__bf16)f1.y; u.e[6]  = (__bf16)f1.z; u.e[7]  = (__bf16)f1.w;
  u.e[8]  = (__bf16)f2.x; u.e[9]  = (__bf16)f2.y; u.e[10] = (__bf16)f2.z; u.e[11] = (__bf16)f2.w;
  u.e[12] = (__bf16)f3.x; u.e[13] = (__bf16)f3.y; u.e[14] = (__bf16)f3.z; u.e[15] = (__bf16)f3.w;
  return u.v;
}

static __device__ inline v8f wmma_bf16(v16bf a, v16bf b, v8f c) {
  return __builtin_amdgcn_wmma_f32_16x16x32_bf16(
      /*neg_a=*/false, a, /*neg_b=*/false, b,
      /*c_mod=*/(short)0, c, /*reuse_a=*/false, /*reuse_b=*/false);
}

// ------------------------------ prep kernels -------------------------------

__global__ void zero_f32(float* p, long n) {
  long i = (long)blockIdx.x * blockDim.x + threadIdx.x;
  long s = (long)gridDim.x * blockDim.x;
  for (; i < n; i += s) p[i] = 0.0f;
}

__global__ void cvt_f32_bf16(const float* __restrict__ s, bf16_t* __restrict__ d, long n) {
  long i = (long)blockIdx.x * blockDim.x + threadIdx.x;
  long st = (long)gridDim.x * blockDim.x;
  for (; i < n; i += st) d[i] = (bf16_t)s[i];
}

// wt[n*K + k] = (bf16) w[k*N + n]   (transpose so B-operand loads are b128s)
__global__ void cvt_transpose(const float* __restrict__ w, bf16_t* __restrict__ wt,
                              int K, int N) {
  long total = (long)K * N;
  long i = (long)blockIdx.x * blockDim.x + threadIdx.x;
  long st = (long)gridDim.x * blockDim.x;
  for (; i < total; i += st) {
    int n = (int)(i / K);
    int k = (int)(i % K);
    wt[i] = (bf16_t)w[(long)k * N + n];
  }
}

// ------------------------------ edge kernel --------------------------------
// One wave per MT*16 = 32 edges. E/32 = 25000 tiles exactly.

__global__ __launch_bounds__(32) void edge_mlp_scatter(
    const bf16_t* __restrict__ xb, const int* __restrict__ eidx,
    const bf16_t* __restrict__ wt0, const float* __restrict__ b0,
    const bf16_t* __restrict__ wt1, const float* __restrict__ b1,
    float* __restrict__ agg)
{
  __shared__ __align__(16) bf16_t lds[MT * 16 * DHID];   // 16 KB
  const int lane = threadIdx.x;
  const int m = lane & 15, g = lane >> 4;
  const long tb = (long)blockIdx.x * (16 * MT);

  // A: MT x (16x256) = [x[row] || x[col]] in bf16 A-layout
  v16bf A[MT][8];
  #pragma unroll
  for (int i = 0; i < MT; ++i) {
    const int nrow = eidx[tb + 16 * i + m];
    const int ncol = eidx[(long)EDGES + tb + 16 * i + m];
    const bf16_t* xr = xb + (long)nrow * DIN;
    const bf16_t* xc = xb + (long)ncol * DIN;
    #pragma unroll
    for (int c = 0; c < 4; ++c) {
      A[i][c]     = load_bf_chunk(xr, c * 32, g);
      A[i][4 + c] = load_bf_chunk(xc, c * 32, g);
    }
  }

  // ---- layer e0: (MT*16 x 256) @ (256x256) + b0, softplus -> LDS ----
  for (int t = 0; t < 16; ++t) {
    const int n = t * 16 + m;                 // C/D layout: N = lane%16
    const float bias = b0[n];
    v8f acc[MT];
    #pragma unroll
    for (int i = 0; i < MT; ++i)
      #pragma unroll
      for (int j = 0; j < 8; ++j) acc[i][j] = bias;
    const bf16_t* wcol = wt0 + (long)n * (2 * DIN);
    #pragma unroll
    for (int c = 0; c < 8; ++c) {
      const v16bf Bc = load_bf_chunk(wcol, c * 32, g);   // loaded once, used MT x
      #pragma unroll
      for (int i = 0; i < MT; ++i) acc[i] = wmma_bf16(A[i][c], Bc, acc[i]);
    }
    #pragma unroll
    for (int i = 0; i < MT; ++i)
      #pragma unroll
      for (int j = 0; j < 8; ++j)             // C/D layout: M = 8g + j
        lds[(i * 16 + 8 * g + j) * DHID + n] = (bf16_t)softplus_f(acc[i][j]);
  }
  __syncthreads();

  // restage layer-0 output back into A-layout (contiguous b128 LDS reads)
  #pragma unroll
  for (int i = 0; i < MT; ++i)
    #pragma unroll
    for (int c = 0; c < 8; ++c)
      A[i][c] = load_bf_chunk(&lds[(i * 16 + m) * DHID], c * 32, g);

  // row nodes for the scatter rows M = 8g..8g+7 of each tile
  int rn[MT][8];
  #pragma unroll
  for (int i = 0; i < MT; ++i) {
    const int4* ep = reinterpret_cast<const int4*>(eidx + tb + 16 * i + 8 * g);
    int4 r0 = ep[0], r1 = ep[1];
    rn[i][0] = r0.x; rn[i][1] = r0.y; rn[i][2] = r0.z; rn[i][3] = r0.w;
    rn[i][4] = r1.x; rn[i][5] = r1.y; rn[i][6] = r1.z; rn[i][7] = r1.w;
  }

  // ---- layer e1: (MT*16 x 256) @ (256x256) + b1, softplus -> atomic scatter
  for (int t = 0; t < 16; ++t) {
    const int n = t * 16 + m;
    const float bias = b1[n];
    v8f acc[MT];
    #pragma unroll
    for (int i = 0; i < MT; ++i)
      #pragma unroll
      for (int j = 0; j < 8; ++j) acc[i][j] = bias;
    const bf16_t* wcol = wt1 + (long)n * DHID;
    #pragma unroll
    for (int c = 0; c < 8; ++c) {
      const v16bf Bc = load_bf_chunk(wcol, c * 32, g);
      #pragma unroll
      for (int i = 0; i < MT; ++i) acc[i] = wmma_bf16(A[i][c], Bc, acc[i]);
    }
    #pragma unroll
    for (int i = 0; i < MT; ++i)
      #pragma unroll
      for (int j = 0; j < 8; ++j)
        atomicAdd(&agg[(long)rn[i][j] * DHID + n], softplus_f(acc[i][j]));
  }
}

// ------------------------------ node kernel --------------------------------
// One wave per 16-node tile. BN/16 = 3125 tiles exactly.

__global__ __launch_bounds__(32) void node_mlp(
    const bf16_t* __restrict__ xb, const float* __restrict__ agg,
    const bf16_t* __restrict__ wt0, const float* __restrict__ b0,
    const bf16_t* __restrict__ wt1, const float* __restrict__ b1,
    const bf16_t* __restrict__ wt2, const float* __restrict__ b2,
    float* __restrict__ out)
{
  __shared__ __align__(16) bf16_t lds[16 * DHID];
  const int lane = threadIdx.x;
  const int m = lane & 15, g = lane >> 4;
  const long node = (long)blockIdx.x * 16 + m;

  // A: 16x384 = [x (bf16) || agg (f32->bf16)]  (12 chunks)
  v16bf A[12];
  #pragma unroll
  for (int c = 0; c < 4; ++c) A[c] = load_bf_chunk(xb + node * DIN, c * 32, g);
  #pragma unroll
  for (int c = 0; c < 8; ++c) A[4 + c] = load_f32_chunk(agg + node * DHID, c * 32, g);

  // ---- layer n0: (16x384) @ (384x256) + b0, softplus -> LDS ----
  for (int t = 0; t < 16; ++t) {
    const int n = t * 16 + m;
    const float bias = b0[n];
    v8f acc;
    #pragma unroll
    for (int j = 0; j < 8; ++j) acc[j] = bias;
    const bf16_t* wcol = wt0 + (long)n * (DIN + DHID);
    #pragma unroll
    for (int c = 0; c < 12; ++c)
      acc = wmma_bf16(A[c], load_bf_chunk(wcol, c * 32, g), acc);
    #pragma unroll
    for (int j = 0; j < 8; ++j)
      lds[(8 * g + j) * DHID + n] = (bf16_t)softplus_f(acc[j]);
  }
  __syncthreads();
  #pragma unroll
  for (int c = 0; c < 8; ++c) A[c] = load_bf_chunk(&lds[m * DHID], c * 32, g);
  __syncthreads();

  // ---- layer n1: (16x256) @ (256x256) + b1, softplus -> LDS ----
  for (int t = 0; t < 16; ++t) {
    const int n = t * 16 + m;
    const float bias = b1[n];
    v8f acc;
    #pragma unroll
    for (int j = 0; j < 8; ++j) acc[j] = bias;
    const bf16_t* wcol = wt1 + (long)n * DHID;
    #pragma unroll
    for (int c = 0; c < 8; ++c)
      acc = wmma_bf16(A[c], load_bf_chunk(wcol, c * 32, g), acc);
    #pragma unroll
    for (int j = 0; j < 8; ++j)
      lds[(8 * g + j) * DHID + n] = (bf16_t)softplus_f(acc[j]);
  }
  __syncthreads();
  #pragma unroll
  for (int c = 0; c < 8; ++c) A[c] = load_bf_chunk(&lds[m * DHID], c * 32, g);

  // ---- layer n2: (16x256) @ (256x128) + b2 (no activation) -> out ----
  for (int t = 0; t < 8; ++t) {
    const int n = t * 16 + m;
    const float bias = b2[n];
    v8f acc;
    #pragma unroll
    for (int j = 0; j < 8; ++j) acc[j] = bias;
    const bf16_t* wcol = wt2 + (long)n * DHID;
    #pragma unroll
    for (int c = 0; c < 8; ++c)
      acc = wmma_bf16(A[c], load_bf_chunk(wcol, c * 32, g), acc);
    #pragma unroll
    for (int j = 0; j < 8; ++j)
      out[((long)blockIdx.x * 16 + 8 * g + j) * DOUT + n] = acc[j];
  }
}

// ------------------------------ launcher -----------------------------------

extern "C" void kernel_launch(void* const* d_in, const int* in_sizes, int n_in,
                              void* d_out, int out_size, void* d_ws, size_t ws_size,
                              hipStream_t stream) {
  (void)in_sizes; (void)n_in; (void)out_size; (void)ws_size;
  const float* x    = (const float*)d_in[0];
  const int*   eidx = (const int*)d_in[1];
  const float* w_e0 = (const float*)d_in[2];
  const float* b_e0 = (const float*)d_in[3];
  const float* w_e1 = (const float*)d_in[4];
  const float* b_e1 = (const float*)d_in[5];
  const float* w_n0 = (const float*)d_in[6];
  const float* b_n0 = (const float*)d_in[7];
  const float* w_n1 = (const float*)d_in[8];
  const float* b_n1 = (const float*)d_in[9];
  const float* w_n2 = (const float*)d_in[10];
  const float* b_n2 = (const float*)d_in[11];
  float* out = (float*)d_out;

  // Workspace carve-out (~64.6 MB): bf16 x, transposed bf16 weights, f32 agg.
  char* ws = (char*)d_ws;
  size_t off = 0;
  auto carve = [&](size_t bytes) -> char* {
    off = (off + 255) & ~(size_t)255;
    char* p = ws + off;
    off += bytes;
    return p;
  };
  bf16_t* xb   = (bf16_t*)carve((size_t)BNODES * DIN * 2);
  bf16_t* wte0 = (bf16_t*)carve((size_t)(2 * DIN) * DHID * 2);
  bf16_t* wte1 = (bf16_t*)carve((size_t)DHID * DHID * 2);
  bf16_t* wtn0 = (bf16_t*)carve((size_t)(DIN + DHID) * DHID * 2);
  bf16_t* wtn1 = (bf16_t*)carve((size_t)DHID * DHID * 2);
  bf16_t* wtn2 = (bf16_t*)carve((size_t)DHID * DOUT * 2);
  float*  agg  = (float*)carve((size_t)BNODES * DHID * 4);

  zero_f32<<<2048, 256, 0, stream>>>(agg, (long)BNODES * DHID);
  cvt_f32_bf16<<<2048, 256, 0, stream>>>(x, xb, (long)BNODES * DIN);
  cvt_transpose<<<256, 256, 0, stream>>>(w_e0, wte0, 2 * DIN, DHID);
  cvt_transpose<<<256, 256, 0, stream>>>(w_e1, wte1, DHID, DHID);
  cvt_transpose<<<256, 256, 0, stream>>>(w_n0, wtn0, DIN + DHID, DHID);
  cvt_transpose<<<256, 256, 0, stream>>>(w_n1, wtn1, DHID, DHID);
  cvt_transpose<<<256, 256, 0, stream>>>(w_n2, wtn2, DHID, DOUT);

  edge_mlp_scatter<<<EDGES / (16 * MT), 32, 0, stream>>>(xb, eidx, wte0, b_e0,
                                                         wte1, b_e1, agg);
  node_mlp<<<BNODES / 16, 32, 0, stream>>>(xb, agg, wtn0, b_n0, wtn1, b_n1,
                                           wtn2, b_n2, out);
}
